// BiLSTM_CRF_31215822308002
// MI455X (gfx1250) — matrix operand order repaired and checked by reading, not compiled, not run on hardware
//
#include <hip/hip_runtime.h>
#include <hip/hip_bf16.h>
#include <math.h>

typedef __bf16 bf16_t;
typedef __attribute__((ext_vector_type(8)))  bf16_t v8bf;
typedef __attribute__((ext_vector_type(16))) bf16_t v16bf;
typedef __attribute__((ext_vector_type(8)))  float  v8f;

#define NB   64
#define NT   512
#define NE   256
#define NH   256
#define NHD  512
#define NG   1024   // 4*H gates
#define NKT  20     // tags
#define CRF_START 18
#define CRF_STOP  19
#define NEGV -10000.0f
#define LDH  264    // padded LDS row: 256 + 8 bf16 (16B) -> conflict-free frag reads
#define WLD  264    // padded LDS row for staged weight panel (k-chunk = 256)

static __device__ __forceinline__ int flip_t(int t, int len) {
  return t < len ? (len - 1 - t) : t;
}
static __device__ __forceinline__ float sigf(float x) {
  return 1.0f / (1.0f + __expf(-x));
}
static __device__ __forceinline__ v16bf join8(v8bf lo, v8bf hi) {
  return __builtin_shufflevector(lo, hi, 0,1,2,3,4,5,6,7,8,9,10,11,12,13,14,15);
}

// WMMA wrapper: RA/RB are the ISA reuse-A/B operand-cache hints (compile-time).
template <bool RA, bool RB>
static __device__ __forceinline__ v8f wmma_bf16(v16bf a, v16bf b, v8f c) {
  return __builtin_amdgcn_wmma_f32_16x16x32_bf16(false, a, false, b,
                                                 (short)0, c, RA, RB);
}

// ---------------------------------------------------------------------------
// float -> bf16 conversion (weights staged once)
// ---------------------------------------------------------------------------
__global__ void cvt_bf16_kernel(const float* __restrict__ src,
                                bf16_t* __restrict__ dst, int n) {
  int i = blockIdx.x * blockDim.x + threadIdx.x;
  if (i < n) dst[i] = (bf16_t)src[i];
}

// ---------------------------------------------------------------------------
// Embedding gather: x0[b*T+t][0:E] = bf16(emb[sent[b*T+t]])
// ---------------------------------------------------------------------------
__global__ void gather_embed_kernel(const float* __restrict__ emb,
                                    const int* __restrict__ sent,
                                    bf16_t* __restrict__ x0) {
  const int rowid = blockIdx.x;            // b*T + t
  const int tok = sent[rowid];
  const float* e = emb + (size_t)tok * NE;
  bf16_t* o = x0 + (size_t)rowid * NE;
  for (int i = threadIdx.x; i < NE; i += blockDim.x) o[i] = (bf16_t)e[i];
}

// ---------------------------------------------------------------------------
// Input-projection GEMM (WMMA bf16, fp32 acc):
//   out[t*64+b][n] = sum_k X[b][src_t][k] * W[n][k] + bias[n]
// The 64xK weight panel for the block is staged into LDS in 64x256 chunks
// (all 8 waves share it); WMMA B-fragments then come from ds_read_b128.
// flip!=0 gathers the time-reversed sequence (packed-seq emulation).
// Block: 256 thr = 8 waves; block tile = 128 rows x 64 cols; wave = 16x64.
// ---------------------------------------------------------------------------
__global__ void __launch_bounds__(256)
gemm_xpre_kernel(const bf16_t* __restrict__ X, int Kdim,
                 const bf16_t* __restrict__ W,
                 const float* __restrict__ bias,
                 float* __restrict__ out,
                 const int* __restrict__ lens, int flip) {
  __shared__ bf16_t wlds[64 * WLD];

  const int tid  = threadIdx.x;
  const int wave = tid >> 5, lane = tid & 31;
  const int l15  = lane & 15;
  const int off8 = (lane & 16) ? 8 : 0;
  const int ro_base = blockIdx.y * 128 + wave * 16;   // output row (t*64+b)
  const int n0 = blockIdx.x * 64;

  // per-lane A row: decode (t,b), apply optional flip
  const int rl = ro_base + l15;
  const int t = rl >> 6, b = rl & 63;
  int src = t;
  if (flip) src = flip_t(t, lens[b]);
  const bf16_t* arow = X + ((size_t)b * NT + src) * (size_t)Kdim;

  v8f zero = {0.f,0.f,0.f,0.f,0.f,0.f,0.f,0.f};
  v8f acc[4];
  acc[0] = zero; acc[1] = zero; acc[2] = zero; acc[3] = zero;

#define GEMM_TILE(nt, RA)                                                 \
  {                                                                       \
    const bf16_t* wrow = &wlds[((nt) * 16 + l15) * WLD + kloc];           \
    v16bf bb = join8(*(const v8bf*)wrow, *(const v8bf*)(wrow + 16));      \
    acc[nt] = wmma_bf16<RA, false>(a, bb, acc[nt]);                       \
  }

  const int nchunks = Kdim >> 8;            // K handled in chunks of 256
  for (int kc = 0; kc < nchunks; ++kc) {
    const int kbase = kc << 8;
    // cooperative stage: W[n0..n0+63][kbase..kbase+255] -> LDS
    for (int idx = tid; idx < 64 * 32; idx += 256) {
      const int rr = idx >> 5, cc = (idx & 31) << 3;
      *(v8bf*)&wlds[rr * WLD + cc] =
          *(const v8bf*)(W + (size_t)(n0 + rr) * Kdim + kbase + cc);
    }
    __syncthreads();
    #pragma unroll
    for (int kk = 0; kk < 8; ++kk) {
      const int kloc = (kk << 5) + off8;
      v16bf a = join8(*(const v8bf*)(arow + kbase + kloc),
                      *(const v8bf*)(arow + kbase + kloc + 16));
      GEMM_TILE(0, true)
      GEMM_TILE(1, true)
      GEMM_TILE(2, true)
      GEMM_TILE(3, false)
    }
    __syncthreads();
  }
#undef GEMM_TILE

  const int hi8 = (lane >> 4) << 3;
  #pragma unroll
  for (int nt = 0; nt < 4; ++nt) {
    const int n = n0 + nt * 16 + l15;
    const float bv = bias[n];
    #pragma unroll
    for (int r = 0; r < 8; ++r) {
      const int ro = ro_base + r + hi8;
      out[(size_t)ro * NG + n] = acc[nt][r] + bv;
    }
  }
}

// ---------------------------------------------------------------------------
// Persistent LSTM scan (one workgroup per direction; grid.x = 2).
// Per step: gates[64x1024] = h[64x256] @ WhhT (WMMA bf16, A from LDS) + x_pre,
// then i/f/g/o nonlinearity fully in registers (c stays in VGPRs).
// Wave layout: 32 waves; mtile=wave>>3 picks 16 batch rows; wq=wave&7 picks
// two hidden-column tiles jt = {2wq, 2wq+1}; each jt covers its i/f/g/o
// n-tiles {jt, jt+16, jt+32, jt+48} so one wave owns all 4 gates of its cells.
// Next step's 256KB x_pre block is prefetched (global_prefetch) each step.
// ---------------------------------------------------------------------------
__global__ void __launch_bounds__(1024)
lstm_scan_kernel(const float* __restrict__ xpre_f,
                 const float* __restrict__ xpre_b,
                 const bf16_t* __restrict__ whh_f,
                 const bf16_t* __restrict__ whh_b,
                 const int* __restrict__ lens,
                 float* __restrict__ hout_f32,
                 bf16_t* __restrict__ hout_bf) {
  __shared__ bf16_t h_lds[64 * LDH];

  const int dir = blockIdx.x;
  const float*  xpre = dir ? xpre_b : xpre_f;
  const bf16_t* whh  = dir ? whh_b  : whh_f;
  const int coloff = dir * NH;

  const int tid  = threadIdx.x;
  const int wave = tid >> 5, lane = tid & 31;
  const int l15  = lane & 15;
  const int off8 = (lane & 16) ? 8 : 0;
  const int hi8  = (lane >> 4) << 3;
  const int m0   = (wave >> 3) * 16;   // batch-row tile
  const int wq   = wave & 7;           // column-tile pair selector
  const int jtv[2] = { wq * 2, wq * 2 + 1 };

  for (int i = tid; i < 64 * LDH; i += 1024) h_lds[i] = (bf16_t)0.0f;

  float creg[2][8];
  int   len_r[8];
  #pragma unroll
  for (int r = 0; r < 8; ++r) {
    len_r[r] = lens[m0 + r + hi8];
    creg[0][r] = 0.f; creg[1][r] = 0.f;
  }

  __syncthreads();

  v8f zero = {0.f,0.f,0.f,0.f,0.f,0.f,0.f,0.f};

#define LSTM_TILE(p, g, RA)                                                  \
  {                                                                          \
    const bf16_t* wrow =                                                     \
        whh + (size_t)((g) * NH + jtv[p] * 16 + l15) * NH + k0;              \
    v16bf bb = join8(*(const v8bf*)wrow, *(const v8bf*)(wrow + 16));         \
    acc[p][g] = wmma_bf16<RA, false>(a, bb, acc[p][g]);                      \
  }

  for (int t = 0; t < NT; ++t) {
    // hide HBM latency of the serial chain: prefetch next step's x_pre block
    if (t + 1 < NT) {
      const char* pf = (const char*)(xpre + (size_t)(t + 1) * 64 * NG);
      __builtin_prefetch(pf + (size_t)tid * 256, 0, 3);
      __builtin_prefetch(pf + (size_t)tid * 256 + 128, 0, 3);
    }

    v8f acc[2][4];
    #pragma unroll
    for (int p = 0; p < 2; ++p)
      #pragma unroll
      for (int g = 0; g < 4; ++g) acc[p][g] = zero;

    // gates += h @ Whh^T  (A from LDS, B streamed through L2)
    #pragma unroll 2
    for (int kk = 0; kk < 8; ++kk) {
      const int k0 = (kk << 5) + off8;
      const bf16_t* hrow = &h_lds[(m0 + l15) * LDH + k0];
      v16bf a = join8(*(const v8bf*)hrow, *(const v8bf*)(hrow + 16));
      LSTM_TILE(0, 0, true)
      LSTM_TILE(0, 1, true)
      LSTM_TILE(0, 2, true)
      LSTM_TILE(0, 3, true)
      LSTM_TILE(1, 0, true)
      LSTM_TILE(1, 1, true)
      LSTM_TILE(1, 2, true)
      LSTM_TILE(1, 3, false)
    }
    __syncthreads();   // all reads of old h complete

    // nonlinearity on WMMA C-layout: lane owns (m = m0+r+hi8, j = jt*16+l15)
    #pragma unroll
    for (int p = 0; p < 2; ++p) {
      const int j = jtv[p] * 16 + l15;
      #pragma unroll
      for (int r = 0; r < 8; ++r) {
        const int m = m0 + r + hi8;
        const size_t base = ((size_t)t * 64 + m) * NG;
        const float iv = xpre[base +          j] + acc[p][0][r];
        const float fv = xpre[base + NH     + j] + acc[p][1][r];
        const float gv = xpre[base + 2 * NH + j] + acc[p][2][r];
        const float ov = xpre[base + 3 * NH + j] + acc[p][3][r];
        const float c = sigf(fv) * creg[p][r] + sigf(iv) * tanhf(gv);
        creg[p][r] = c;
        const float h = sigf(ov) * tanhf(c);
        h_lds[m * LDH + j] = (bf16_t)h;

        const int len = len_r[r];
        const int tout = dir ? flip_t(t, len) : t;
        const float hm = (t < len) ? h : 0.f;
        const size_t oidx = ((size_t)m * NT + tout) * NHD + coloff + j;
        if (hout_bf)  hout_bf[oidx]  = (bf16_t)hm;
        if (hout_f32) hout_f32[oidx] = hm;
      }
    }
    __syncthreads();   // new h visible before next step's reads
  }
#undef LSTM_TILE
}

// ---------------------------------------------------------------------------
// LayerNorm(HD=512) + 20-way output projection (tiny; plain VALU)
// ---------------------------------------------------------------------------
__global__ void __launch_bounds__(128)
ln_feats_kernel(const float* __restrict__ x, const float* __restrict__ gamma,
                const float* __restrict__ beta, const float* __restrict__ w_out,
                const float* __restrict__ b_out, float* __restrict__ feats) {
  __shared__ float xs[NHD];
  __shared__ float red[128], red2[128];
  const int bt = blockIdx.x, tid = threadIdx.x;
  const float* row = x + (size_t)bt * NHD;

  float s = 0.f, s2 = 0.f;
  for (int h = tid; h < NHD; h += 128) {
    const float v = row[h];
    xs[h] = v; s += v; s2 += v * v;
  }
  red[tid] = s; red2[tid] = s2;
  __syncthreads();
  for (int w = 64; w > 0; w >>= 1) {
    if (tid < w) { red[tid] += red[tid + w]; red2[tid] += red2[tid + w]; }
    __syncthreads();
  }
  const float mu = red[0] * (1.0f / NHD);
  const float var = red2[0] * (1.0f / NHD) - mu * mu;
  const float rstd = rsqrtf(var + 1e-5f);
  for (int h = tid; h < NHD; h += 128)
    xs[h] = (xs[h] - mu) * rstd * gamma[h] + beta[h];
  __syncthreads();

  const int wave = tid >> 5, lane = tid & 31;
  for (int k = wave; k < NKT; k += 4) {
    float d = 0.f;
    for (int h = lane; h < NHD; h += 32) d += xs[h] * w_out[k * NHD + h];
    for (int o = 16; o > 0; o >>= 1) d += __shfl_down(d, o, 32);
    if (lane == 0) feats[(size_t)bt * NKT + k] = d + b_out[k];
  }
}

// ---------------------------------------------------------------------------
// CRF forward scan + gold score + mean loss accumulation (one wave per b)
// ---------------------------------------------------------------------------
__global__ void __launch_bounds__(32)
crf_kernel(const float* __restrict__ feats, const float* __restrict__ trans,
           const int* __restrict__ tags, const int* __restrict__ lens,
           float* __restrict__ out) {
  __shared__ float tr[NKT * NKT];
  __shared__ float alpha[NKT];
  const int b = blockIdx.x, lane = threadIdx.x;

  for (int i = lane; i < NKT * NKT; i += 32) tr[i] = trans[i];
  if (lane < NKT) alpha[lane] = (lane == CRF_START) ? 0.f : NEGV;
  __syncthreads();

  const int len = lens[b];
  const float* frow = feats + (size_t)b * NT * NKT;

  for (int t = 0; t < len; ++t) {
    float nv = 0.f;
    if (lane < NKT) {
      float m = -1e30f;
      #pragma unroll
      for (int j = 0; j < NKT; ++j) m = fmaxf(m, alpha[j] + tr[lane * NKT + j]);
      float s = 0.f;
      #pragma unroll
      for (int j = 0; j < NKT; ++j) s += __expf(alpha[j] + tr[lane * NKT + j] - m);
      nv = frow[t * NKT + lane] + m + __logf(s);
    }
    __syncthreads();
    if (lane < NKT) alpha[lane] = nv;
    __syncthreads();
  }

  // logZ = lse_k(alpha_T[k] + trans[STOP][k])
  float v = (lane < NKT) ? (alpha[lane] + tr[CRF_STOP * NKT + lane]) : -1e30f;
  float m = v;
  for (int o = 16; o > 0; o >>= 1) m = fmaxf(m, __shfl_down(m, o, 32));
  m = __shfl(m, 0, 32);
  float e = (lane < NKT) ? __expf(v - m) : 0.f;
  for (int o = 16; o > 0; o >>= 1) e += __shfl_down(e, o, 32);

  if (lane == 0) {
    const float logZ = m + __logf(e);
    float gold = 0.f;
    int prev = CRF_START;
    const int* trow = tags + (size_t)b * NT;
    for (int t = 0; t < len; ++t) {
      const int tg = trow[t];
      gold += tr[tg * NKT + prev] + frow[t * NKT + tg];
      prev = tg;
    }
    gold += tr[CRF_STOP * NKT + prev];
    atomicAdd(out, (logZ - gold) * (1.0f / NB));
  }
}

__global__ void zero_kernel(float* o) {
  if (threadIdx.x == 0 && blockIdx.x == 0) o[0] = 0.f;
}

// ---------------------------------------------------------------------------
extern "C" void kernel_launch(void* const* d_in, const int* in_sizes, int n_in,
                              void* d_out, int out_size, void* d_ws, size_t ws_size,
                              hipStream_t stream) {
  (void)in_sizes; (void)n_in; (void)out_size; (void)ws_size;
  const float* emb   = (const float*)d_in[0];
  const float* wih1f = (const float*)d_in[1];
  const float* whh1f = (const float*)d_in[2];
  const float* b1f   = (const float*)d_in[3];
  const float* wih1b = (const float*)d_in[4];
  const float* whh1b = (const float*)d_in[5];
  const float* b1b   = (const float*)d_in[6];
  const float* wih2f = (const float*)d_in[7];
  const float* whh2f = (const float*)d_in[8];
  const float* b2f   = (const float*)d_in[9];
  const float* wih2b = (const float*)d_in[10];
  const float* whh2b = (const float*)d_in[11];
  const float* b2b   = (const float*)d_in[12];
  const float* gamma = (const float*)d_in[13];
  const float* beta  = (const float*)d_in[14];
  const float* w_out = (const float*)d_in[15];
  const float* b_out = (const float*)d_in[16];
  const float* trans = (const float*)d_in[17];
  const int*   sent  = (const int*)d_in[18];
  const int*   lens  = (const int*)d_in[19];
  const int*   tags  = (const int*)d_in[20];
  float* out = (float*)d_out;

  char* ws = (char*)d_ws;
  size_t off = 0;
  auto alloc = [&](size_t bytes) -> void* {
    void* p = ws + off;
    off += (bytes + 255) & ~(size_t)255;
    return p;
  };

  bf16_t* x0_bf   = (bf16_t*)alloc((size_t)NB * NT * NE  * sizeof(bf16_t));
  bf16_t* x1_bf   = (bf16_t*)alloc((size_t)NB * NT * NHD * sizeof(bf16_t));
  float*  xpre_f  = (float*) alloc((size_t)NT * NB * NG  * sizeof(float));
  float*  xpre_b  = (float*) alloc((size_t)NT * NB * NG  * sizeof(float));
  float*  h2      = (float*) alloc((size_t)NB * NT * NHD * sizeof(float));
  float*  feats   = (float*) alloc((size_t)NB * NT * NKT * sizeof(float));
  bf16_t* wih1f_bf = (bf16_t*)alloc((size_t)NG * NE  * sizeof(bf16_t));
  bf16_t* whh1f_bf = (bf16_t*)alloc((size_t)NG * NH  * sizeof(bf16_t));
  bf16_t* wih1b_bf = (bf16_t*)alloc((size_t)NG * NE  * sizeof(bf16_t));
  bf16_t* whh1b_bf = (bf16_t*)alloc((size_t)NG * NH  * sizeof(bf16_t));
  bf16_t* wih2f_bf = (bf16_t*)alloc((size_t)NG * NHD * sizeof(bf16_t));
  bf16_t* whh2f_bf = (bf16_t*)alloc((size_t)NG * NH  * sizeof(bf16_t));
  bf16_t* wih2b_bf = (bf16_t*)alloc((size_t)NG * NHD * sizeof(bf16_t));
  bf16_t* whh2b_bf = (bf16_t*)alloc((size_t)NG * NH  * sizeof(bf16_t));

  zero_kernel<<<1, 1, 0, stream>>>(out);

  struct Cv { const float* s; bf16_t* d; int n; };
  const Cv cv[8] = {
    {wih1f, wih1f_bf, NG * NE}, {whh1f, whh1f_bf, NG * NH},
    {wih1b, wih1b_bf, NG * NE}, {whh1b, whh1b_bf, NG * NH},
    {wih2f, wih2f_bf, NG * NHD}, {whh2f, whh2f_bf, NG * NH},
    {wih2b, wih2b_bf, NG * NHD}, {whh2b, whh2b_bf, NG * NH},
  };
  for (int i = 0; i < 8; ++i)
    cvt_bf16_kernel<<<(cv[i].n + 255) / 256, 256, 0, stream>>>(cv[i].s, cv[i].d, cv[i].n);

  gather_embed_kernel<<<NB * NT, 256, 0, stream>>>(emb, sent, x0_bf);

  const dim3 ggrid(NG / 64, (NB * NT) / 128);

  // ---- layer 1 ----
  gemm_xpre_kernel<<<ggrid, 256, 0, stream>>>(x0_bf, NE, wih1f_bf, b1f, xpre_f, lens, 0);
  gemm_xpre_kernel<<<ggrid, 256, 0, stream>>>(x0_bf, NE, wih1b_bf, b1b, xpre_b, lens, 1);
  lstm_scan_kernel<<<2, 1024, 0, stream>>>(xpre_f, xpre_b, whh1f_bf, whh1b_bf,
                                           lens, nullptr, x1_bf);

  // ---- layer 2 ----
  gemm_xpre_kernel<<<ggrid, 256, 0, stream>>>(x1_bf, NHD, wih2f_bf, b2f, xpre_f, lens, 0);
  gemm_xpre_kernel<<<ggrid, 256, 0, stream>>>(x1_bf, NHD, wih2b_bf, b2b, xpre_b, lens, 1);
  lstm_scan_kernel<<<2, 1024, 0, stream>>>(xpre_f, xpre_b, whh2f_bf, whh2b_bf,
                                           lens, h2, nullptr);

  // ---- layernorm + feats ----
  ln_feats_kernel<<<NB * NT, 128, 0, stream>>>(h2, gamma, beta, w_out, b_out, feats);

  // ---- CRF loss ----
  crf_kernel<<<NB, 32, 0, stream>>>(feats, trans, tags, lens, out);
}
